// CWS_2250562863727
// MI455X (gfx1250) — matrix operand order repaired and checked by reading, not compiled, not run on hardware
//
#include <hip/hip_runtime.h>
#include <hip/hip_bf16.h>

// ---------------------------------------------------------------------------
// Types for CDNA5 WMMA (wave32)
// ---------------------------------------------------------------------------
typedef __attribute__((ext_vector_type(16))) __bf16        v16bf;
typedef __attribute__((ext_vector_type(8)))  float         v8f;
typedef __attribute__((ext_vector_type(4)))  unsigned int  v4u;

union FragU { struct { v4u lo; v4u hi; } p; v16bf f; };

static __device__ inline v16bf make_frag(v4u lo, v4u hi) {
    FragU u; u.p.lo = lo; u.p.hi = hi; return u.f;
}

static __device__ inline float bf2f(unsigned short h) {
    unsigned int u = ((unsigned int)h) << 16;
    float f; __builtin_memcpy(&f, &u, 4); return f;
}
static __device__ inline unsigned short f2bf(float f) {
    unsigned int u; __builtin_memcpy(&u, &f, 4);
    unsigned int r = u + 0x7FFFu + ((u >> 16) & 1u);
    return (unsigned short)(r >> 16);
}
static __device__ inline float sigm(float x) { return 1.f / (1.f + expf(-x)); }

// ---------------------------------------------------------------------------
// f32 -> bf16 conversion
// ---------------------------------------------------------------------------
__global__ void cvt_bf16_kernel(const float* __restrict__ src,
                                unsigned short* __restrict__ dst, int n) {
    int i = blockIdx.x * blockDim.x + threadIdx.x;
    if (i < n) dst[i] = f2bf(src[i]);
}

// ---------------------------------------------------------------------------
// Embedding gather: x0[bt, e] = bf16(emb[sentence[bt], e])
// ---------------------------------------------------------------------------
__global__ void embed_kernel(const int* __restrict__ sent,
                             const float* __restrict__ emb,
                             unsigned short* __restrict__ x0, int n, int E) {
    int i = blockIdx.x * blockDim.x + threadIdx.x;
    if (i >= n) return;
    int bt = i / E;
    int e  = i - bt * E;
    x0[i] = f2bf(emb[(size_t)sent[bt] * E + e]);
}

// ---------------------------------------------------------------------------
// Per-wave WMMA GEMM:  Out(MxN, bf16) = act( X(MxK, bf16) @ W(NxK, bf16)^T + bias )
// One wave computes a 16x64 strip (4 N-tiles sharing one A fragment);
// 8 waves / block.
// A frag (16-bit 16x32): lane<16 -> K {0..7,16..23}; lane>=16 -> {8..15,24..31}
// B frag              : lane<16 -> col N=lane, K {0..7,16..23}; lane>=16 -> {8..15,24..31}
// ---------------------------------------------------------------------------
__global__ void gemm_bf16_kernel(const unsigned short* __restrict__ X,
                                 const unsigned short* __restrict__ W,
                                 const float* __restrict__ bias,
                                 unsigned short* __restrict__ Out,
                                 int M, int N, int K, int relu) {
    int wave = threadIdx.x >> 5;
    int lane = threadIdx.x & 31;
    int ntN  = N >> 6;                      // 16x64 strips along N
    int tile = blockIdx.x * 8 + wave;
    int mt   = tile / ntN, nt = tile - mt * ntN;
    if (mt >= (M >> 4)) return;
    int m0 = mt << 4, n0 = nt << 6;
    int l15 = lane & 15;
    int hi  = lane >> 4;

    const unsigned short* arow = X + (size_t)(m0 + l15) * K + hi * 8;
    const unsigned short* brow = W + (size_t)(n0 + l15) * K + hi * 8;
    const size_t bstep = (size_t)16 * K;    // next N-tile of W

    v8f acc[4];
    v8f z = {};
#pragma unroll
    for (int j = 0; j < 4; ++j) acc[j] = z;

    for (int k0 = 0; k0 < K; k0 += 32) {
        v4u alo = *(const v4u*)(arow + k0);
        v4u ahi = *(const v4u*)(arow + k0 + 16);
        v16bf af = make_frag(alo, ahi);
#pragma unroll
        for (int j = 0; j < 4; ++j) {
            v4u blo = *(const v4u*)(brow + (size_t)j * bstep + k0);
            v4u bhi = *(const v4u*)(brow + (size_t)j * bstep + k0 + 16);
            acc[j] = __builtin_amdgcn_wmma_f32_16x16x32_bf16(
                false, af, false, make_frag(blo, bhi),
                (short)0, acc[j], false, false);
        }
    }

#pragma unroll
    for (int j = 0; j < 4; ++j) {
        int col = n0 + j * 16 + l15;
        float bv = bias ? bias[col] : 0.f;
        size_t cbase = (size_t)(m0 + hi * 8) * N + col;
#pragma unroll
        for (int r = 0; r < 8; ++r) {
            float v = acc[j][r] + bv;
            if (relu) v = v > 0.f ? v : 0.f;
            Out[cbase + (size_t)r * N] = f2bf(v);
        }
    }
}

// ---------------------------------------------------------------------------
// LSTM recurrence (one layer, both directions in one launch).
// grid = (4 batch-chunks, 2 dirs), block = 256 (8 waves).
// Wave w owns gate columns {g*256 + 32w .. +31} for g=i,f,g,o => the full cell
// update for hidden units [32w, 32w+32) lives in this wave's accumulators.
// h (16x256 bf16) lives in LDS and feeds WMMA A-fragments each step.
// ---------------------------------------------------------------------------
__global__ void lstm_kernel(const unsigned short* __restrict__ xw_f,
                            const unsigned short* __restrict__ xw_b,
                            const unsigned short* __restrict__ Whh_f,
                            const unsigned short* __restrict__ Whh_b,
                            unsigned short* __restrict__ hout, // (B*T) x 512
                            const int* __restrict__ lengths, int T) {
    __shared__ unsigned short hlds[16 * 256];

    const int dir   = blockIdx.y;
    const int chunk = blockIdx.x;
    const unsigned short* xw  = dir ? xw_b  : xw_f;
    const unsigned short* Whh = dir ? Whh_b : Whh_f;

    const int wave = threadIdx.x >> 5;
    const int lane = threadIdx.x & 31;
    const int l15  = lane & 15;
    const int hi   = lane >> 4;

    for (int i = threadIdx.x; i < 16 * 256; i += 256) hlds[i] = 0;

    int lens[8];
#pragma unroll
    for (int r = 0; r < 8; ++r) lens[r] = lengths[chunk * 16 + r + 8 * hi];

    float c[2][8];
#pragma unroll
    for (int p = 0; p < 2; ++p)
#pragma unroll
        for (int r = 0; r < 8; ++r) c[p][r] = 0.f;

    const unsigned short* brow[8];
#pragma unroll
    for (int g = 0; g < 4; ++g)
#pragma unroll
        for (int p = 0; p < 2; ++p)
            brow[g * 2 + p] =
                Whh + (size_t)(g * 256 + wave * 32 + p * 16 + l15) * 256 + hi * 8;

    const unsigned short* ar = &hlds[l15 * 256 + hi * 8];
    __syncthreads();

    for (int s = 0; s < T; ++s) {
        v8f acc[8];
        v8f z = {};
#pragma unroll
        for (int i = 0; i < 8; ++i) acc[i] = z;

        // g += h_prev @ Whh^T   (h from LDS, W from L2-resident global)
        for (int k0 = 0; k0 < 256; k0 += 32) {
            v4u alo = *(const v4u*)(ar + k0);
            v4u ahi = *(const v4u*)(ar + k0 + 16);
            v16bf af = make_frag(alo, ahi);
#pragma unroll
            for (int i = 0; i < 8; ++i) {
                v4u blo = *(const v4u*)(brow[i] + k0);
                v4u bhi = *(const v4u*)(brow[i] + k0 + 16);
                acc[i] = __builtin_amdgcn_wmma_f32_16x16x32_bf16(
                    false, af, false, make_frag(blo, bhi),
                    (short)0, acc[i], false, false);
            }
        }
        __syncthreads();  // everyone done reading h_prev

        // cell update entirely in registers; write h to LDS + global
#pragma unroll
        for (int p = 0; p < 2; ++p) {
            int j = wave * 32 + p * 16 + l15;  // hidden unit
#pragma unroll
            for (int r = 0; r < 8; ++r) {
                int row = r + 8 * hi;
                int bg  = chunk * 16 + row;
                int len = lens[r];
                int tix = (dir && (s < len)) ? (len - 1 - s) : s;
                size_t xb = ((size_t)bg * T + tix) * 1024;
                float gi = acc[0 + p][r] + bf2f(xw[xb + 0   + j]);
                float gf = acc[2 + p][r] + bf2f(xw[xb + 256 + j]);
                float gg = acc[4 + p][r] + bf2f(xw[xb + 512 + j]);
                float go = acc[6 + p][r] + bf2f(xw[xb + 768 + j]);
                float cc = sigm(gf) * c[p][r] + sigm(gi) * tanhf(gg);
                c[p][r]  = cc;
                float hv = sigm(go) * tanhf(cc);
                unsigned short hb = f2bf(hv);
                hlds[row * 256 + j] = hb;
                hout[((size_t)bg * T + tix) * 512 + dir * 256 + j] = hb;
            }
        }
        __syncthreads();  // h published before next step reads it
    }
}

// ---------------------------------------------------------------------------
// emissions[m, n] = b_tag[n] + sum_k h[m,k] * W_tag[n,k]   (NT = 4)
// ---------------------------------------------------------------------------
__global__ void emis_kernel(const unsigned short* __restrict__ h,
                            const float* __restrict__ Wtag,
                            const float* __restrict__ btag,
                            float* __restrict__ emis, int M) {
    int i = blockIdx.x * blockDim.x + threadIdx.x;
    if (i >= M * 4) return;
    int m = i >> 2, n = i & 3;
    const unsigned short* hr = h + (size_t)m * 512;
    const float* wr = Wtag + n * 512;
    float s = btag[n];
    for (int k = 0; k < 512; ++k) s += bf2f(hr[k]) * wr[k];
    emis[i] = s;
}

// ---------------------------------------------------------------------------
// CRF forward + numerator. grid = B, block = 32 (one wave, lanes 0..3 = tags)
// ---------------------------------------------------------------------------
__global__ void crf_kernel(const float* __restrict__ emis,
                           const int* __restrict__ tags,
                           const int* __restrict__ lengths,
                           const float* __restrict__ cstart,
                           const float* __restrict__ cend,
                           const float* __restrict__ ctrans,
                           float* __restrict__ res, int T) {
    int b = blockIdx.x;
    int lane = threadIdx.x;
    int len = lengths[b];
    const float* eb = emis + (size_t)b * T * 4;
    const int*   tb = tags + (size_t)b * T;

    float tr0 = 0.f, tr1 = 0.f, tr2 = 0.f, tr3 = 0.f;
    float alpha = -1e30f;
    if (lane < 4) {
        tr0 = ctrans[0 + lane]; tr1 = ctrans[4 + lane];
        tr2 = ctrans[8 + lane]; tr3 = ctrans[12 + lane];
        alpha = cstart[lane] + eb[lane];
    }
    for (int t = 1; t < T; ++t) {
        float a0 = __shfl(alpha, 0, 32);
        float a1 = __shfl(alpha, 1, 32);
        float a2 = __shfl(alpha, 2, 32);
        float a3 = __shfl(alpha, 3, 32);
        if (lane < 4 && t < len) {
            float v0 = a0 + tr0, v1 = a1 + tr1, v2 = a2 + tr2, v3 = a3 + tr3;
            float m = fmaxf(fmaxf(v0, v1), fmaxf(v2, v3));
            alpha = m + logf(expf(v0 - m) + expf(v1 - m) +
                             expf(v2 - m) + expf(v3 - m)) + eb[t * 4 + lane];
        }
    }
    float fin = (lane < 4) ? alpha + cend[lane] : -1e30f;
    float f0 = __shfl(fin, 0, 32), f1 = __shfl(fin, 1, 32);
    float f2 = __shfl(fin, 2, 32), f3 = __shfl(fin, 3, 32);
    if (lane == 0) {
        float m = fmaxf(fmaxf(f0, f1), fmaxf(f2, f3));
        float logZ = m + logf(expf(f0 - m) + expf(f1 - m) +
                              expf(f2 - m) + expf(f3 - m));
        int tp = tb[0];
        float num = cstart[tp] + eb[tp];
        for (int t = 1; t < T; ++t) {
            int tg = tb[t];
            if (t < len) num += eb[t * 4 + tg] + ctrans[tp * 4 + tg];
            tp = tg;
        }
        num += cend[tb[len - 1]];
        res[b] = num - logZ;
    }
}

__global__ void loss_kernel(const float* __restrict__ res, float* __restrict__ out, int B) {
    __shared__ float sm[64];
    int t = threadIdx.x;
    sm[t] = res[t];
    __syncthreads();
    if (t == 0) {
        float s = 0.f;
        for (int i = 0; i < B; ++i) s += sm[i];
        out[0] = -(s / (float)B);
    }
}

// ---------------------------------------------------------------------------
// Host side
// ---------------------------------------------------------------------------
extern "C" void kernel_launch(void* const* d_in, const int* in_sizes, int n_in,
                              void* d_out, int out_size, void* d_ws, size_t ws_size,
                              hipStream_t stream) {
    (void)in_sizes; (void)n_in; (void)out_size; (void)ws_size;
    const int B = 64, T = 1024, E = 256, NT = 4;
    const size_t M = (size_t)B * T;

    const int*   sent = (const int*)d_in[0];
    const int*   tags = (const int*)d_in[1];
    const int*   lens = (const int*)d_in[3];
    const float* emb  = (const float*)d_in[4];
    // order: f0, b0, f1, b1
    const float* Wih[4]  = { (const float*)d_in[5],  (const float*)d_in[8],
                             (const float*)d_in[11], (const float*)d_in[14] };
    const float* Whh[4]  = { (const float*)d_in[6],  (const float*)d_in[9],
                             (const float*)d_in[12], (const float*)d_in[15] };
    const float* bih[4]  = { (const float*)d_in[7],  (const float*)d_in[10],
                             (const float*)d_in[13], (const float*)d_in[16] };
    const float* Wlin  = (const float*)d_in[17];
    const float* blin  = (const float*)d_in[18];
    const float* Wtag  = (const float*)d_in[19];
    const float* btag  = (const float*)d_in[20];
    const float* cstart = (const float*)d_in[21];
    const float* cend   = (const float*)d_in[22];
    const float* ctrans = (const float*)d_in[23];

    // workspace layout (ushort units)
    unsigned short* base = (unsigned short*)d_ws;
    unsigned short* xwf = base;                 // M*1024 bf16
    unsigned short* xwb = xwf + M * 1024;       // M*1024 bf16
    unsigned short* x0  = xwb + M * 1024;       // M*256  bf16
    unsigned short* l0  = x0  + M * 256;        // M*512  bf16
    unsigned short* l1  = l0  + M * 512;        // M*512  bf16
    unsigned short* p   = l1  + M * 512;
    unsigned short* Wihbf[4]; unsigned short* Whhbf[4];
    const int ihsz[4] = { 1024 * 256, 1024 * 256, 1024 * 512, 1024 * 512 };
    for (int i = 0; i < 4; ++i) {
        Wihbf[i] = p; p += ihsz[i];
        Whhbf[i] = p; p += 1024 * 256;
    }
    unsigned short* Wlinbf = p; p += 512 * 512;
    float* resv = (float*)p;
    unsigned short* hbuf = xwf;        // reuse after LSTM stages
    float* emis = (float*)xwb;         // reuse after LSTM stages

    // weight conversions f32 -> bf16
    for (int i = 0; i < 4; ++i) {
        cvt_bf16_kernel<<<(ihsz[i] + 255) / 256, 256, 0, stream>>>(Wih[i], Wihbf[i], ihsz[i]);
        cvt_bf16_kernel<<<(1024 * 256 + 255) / 256, 256, 0, stream>>>(Whh[i], Whhbf[i], 1024 * 256);
    }
    cvt_bf16_kernel<<<(512 * 512 + 255) / 256, 256, 0, stream>>>(Wlin, Wlinbf, 512 * 512);

    // embedding gather
    {
        int n = (int)(M * E);
        embed_kernel<<<(n + 255) / 256, 256, 0, stream>>>(sent, emb, x0, n, E);
    }

    // layer 0: input projections + recurrence
    gemm_bf16_kernel<<<(int)((M / 16) * (1024 / 64) / 8), 256, 0, stream>>>(
        x0, Wihbf[0], bih[0], xwf, (int)M, 1024, 256, 0);
    gemm_bf16_kernel<<<(int)((M / 16) * (1024 / 64) / 8), 256, 0, stream>>>(
        x0, Wihbf[1], bih[1], xwb, (int)M, 1024, 256, 0);
    lstm_kernel<<<dim3(4, 2), 256, 0, stream>>>(xwf, xwb, Whhbf[0], Whhbf[1], l0, lens, T);

    // layer 1
    gemm_bf16_kernel<<<(int)((M / 16) * (1024 / 64) / 8), 256, 0, stream>>>(
        l0, Wihbf[2], bih[2], xwf, (int)M, 1024, 512, 0);
    gemm_bf16_kernel<<<(int)((M / 16) * (1024 / 64) / 8), 256, 0, stream>>>(
        l0, Wihbf[3], bih[3], xwb, (int)M, 1024, 512, 0);
    lstm_kernel<<<dim3(4, 2), 256, 0, stream>>>(xwf, xwb, Whhbf[2], Whhbf[3], l1, lens, T);

    // linear + relu (output reuses xwf)
    gemm_bf16_kernel<<<(int)((M / 16) * (512 / 64) / 8), 256, 0, stream>>>(
        l1, Wlinbf, blin, hbuf, (int)M, 512, 512, 1);

    // tag projection (NT=4, plain VALU)
    emis_kernel<<<(int)((M * NT + 255) / 256), 256, 0, stream>>>(hbuf, Wtag, btag, emis, (int)M);

    // CRF per batch element + final loss
    crf_kernel<<<B, 32, 0, stream>>>(emis, tags, lens, cstart, cend, ctrans, resv, T);
    loss_kernel<<<1, 64, 0, stream>>>(resv, (float*)d_out, B);
}
